// NestedLSTMCell_15865609191568
// MI455X (gfx1250) — compile-verified, matrix operand order checked by reading, not run on hardware
//
#include <hip/hip_runtime.h>
#include <hip/hip_bf16.h>
#include <stdint.h>

// ---------------------------------------------------------------------------
// Nested LSTM (DEPTH=2, U=1024, T=512, B=64, D_IN=1024) for gfx1250 (MI455X).
//
// Per step: two GEMMs [64 x 2048] x [2048 x 4096] fused with the LSTM gate
// nonlinearities.  Weights converted once per call to bf16 and packed into the
// CDNA5 WMMA B-fragment layout; GEMM uses v_wmma_f32_16x16x32_bf16 with each
// wave owning one (m-tile, u-tile) and 4 gate accumulators (i,c,f,o) that
// share a single A fragment.  Activation chunks are staged into LDS with the
// gfx1250 async global->LDS path (ASYNCcnt), weights are prefetched with
// WGP-scope global_prefetch_b8.
// ---------------------------------------------------------------------------

typedef __attribute__((ext_vector_type(16))) __bf16 v16bf;
typedef __attribute__((ext_vector_type(8)))  float  v8f;

__device__ __forceinline__ unsigned short f2bf(float f) {
    union { float f; unsigned u; } v; v.f = f;
    unsigned u = v.u;
    unsigned r = u + 0x7FFFu + ((u >> 16) & 1u);   // round-to-nearest-even
    return (unsigned short)(r >> 16);
}

__device__ __forceinline__ float sigm(float x) {
    return 1.0f / (1.0f + __expf(-x));
}

// --- fp32 -> bf16 bit-pattern copy (x, h0) ---------------------------------
__global__ void f2bf_copy_kernel(const float* __restrict__ src,
                                 unsigned short* __restrict__ dst, int n) {
    for (int i = blockIdx.x * blockDim.x + threadIdx.x; i < n;
         i += gridDim.x * blockDim.x)
        dst[i] = f2bf(src[i]);
}

// --- fp32 state copy (c1, c2) ----------------------------------------------
__global__ void fcopy_kernel(const float* __restrict__ src,
                             float* __restrict__ dst, int n) {
    for (int i = blockIdx.x * blockDim.x + threadIdx.x; i < n;
         i += gridDim.x * blockDim.x)
        dst[i] = src[i];
}

// --- pack W (2048 x 4096 fp32, row-major) into bf16 WMMA B-fragments -------
// Layout: wp[(((ut*64 + kt)*4 + gate)*32 + lane)*16 + j]
//   lane<16 : n = lane,    k = kt*32 + (j<8 ? j      : j+8 )   // {0..7,16..23}
//   lane>=16: n = lane-16, k = kt*32 + (j<8 ? j+8    : j+16)   // {8..15,24..31}
__global__ void pack_weights_kernel(const float* __restrict__ W,
                                    unsigned short* __restrict__ wp, int total) {
    for (int e = blockIdx.x * blockDim.x + threadIdx.x; e < total;
         e += gridDim.x * blockDim.x) {
        int j    =  e        & 15;
        int lane = (e >> 4)  & 31;
        int gate = (e >> 9)  & 3;
        int kt   = (e >> 11) & 63;
        int ut   =  e >> 17;
        int n = lane & 15;
        int kk = (lane < 16) ? ((j < 8) ? j : j + 8)
                             : ((j < 8) ? j + 8 : j + 16);
        kk += kt * 32;
        wp[e] = f2bf(W[(size_t)kk * 4096 + (size_t)gate * 1024 + ut * 16 + n]);
    }
}

// --- fused GEMM + gate nonlinearity, one time step, one layer --------------
#define BK   128
#define LDA  136   // ushort stride, 272B (16B-aligned, bank-conflict padded)

__global__ __launch_bounds__(256)
void lstm_gemm_step(const unsigned short* __restrict__ wpack,
                    const unsigned short* __restrict__ a_lo,  // cols [0,1024)
                    const unsigned short* __restrict__ a_hi,  // cols [1024,2048)
                    int a_stride_lo, int a_stride_hi,
                    const float* __restrict__ bias,
                    float* __restrict__ c_state,   // L0: c1 (ro) | L1: c2 (rw)
                    float* __restrict__ o0sig,     // L0: write   | L1: read
                    unsigned short* __restrict__ a1bf,  // L0: write [ii|ih]
                    float* __restrict__ c1_state,  // L1: write new_h1
                    float* __restrict__ out_t,     // L1: write h_seq[t]
                    unsigned short* __restrict__ hbf,   // L1: write bf16(h)
                    int layer) {
    __shared__ unsigned short As[64 * LDA];

    const int tid  = threadIdx.x;
    const int lane = tid & 31;
    const int wv   = tid >> 5;
    const int tile = blockIdx.x * 8 + wv;   // 0..255
    const int mt   = tile & 3;              // 4 m-tiles (B=64)
    const int ut   = tile >> 2;             // 64 u-tiles (U=1024)

    v8f acc[4] = {};                        // i, c, f, o accumulators

    const int mrow = mt * 16 + (lane & 15);
    const int offA = (lane < 16) ? 0 : 8;
    const int offB = (lane < 16) ? 16 : 24;

    for (int kc = 0; kc < 16; ++kc) {       // K = 2048, chunks of 128
        // ---- async-stage 64 x 128 bf16 activation chunk into LDS --------
        const int kcbase = kc * BK;
        const unsigned short* src;
        int stride, colofs;
        if (kcbase < 1024) { src = a_lo; stride = a_stride_lo; colofs = kcbase; }
        else               { src = a_hi; stride = a_stride_hi; colofs = kcbase - 1024; }
#pragma unroll
        for (int q = 0; q < 4; ++q) {
            int idx = tid + q * 256;        // 1024 x 16B transfers
            int row = idx >> 4;
            int c8  = (idx & 15) * 8;
            unsigned lds_b = (unsigned)(uintptr_t)(&As[row * LDA + c8]);
            const unsigned short* gp = &src[(size_t)row * stride + colofs + c8];
            asm volatile("global_load_async_to_lds_b128 %0, %1, off"
                         :: "v"(lds_b), "v"(gp) : "memory");
        }
        asm volatile("s_wait_asynccnt 0x0" ::: "memory");
        __syncthreads();

        // ---- 4 k-steps of 32, 4 gates each ------------------------------
#pragma unroll
        for (int kt = 0; kt < 4; ++kt) {
            const int ktg = kc * 4 + kt;
            union { v16bf v; uint4 q[2]; } af;
            const int kb = kt * 32;
            af.q[0] = *(const uint4*)(&As[mrow * LDA + kb + offA]);
            af.q[1] = *(const uint4*)(&As[mrow * LDA + kb + offB]);

            const unsigned short* bp =
                wpack + ((((size_t)ut * 64 + ktg) * 4) * 32 + lane) * 16;
            // prefetch next k-step's weight block (near-temporal, WGP scope)
            __builtin_prefetch(bp + 2048, 0, 3);
#pragma unroll
            for (int g = 0; g < 4; ++g) {
                v16bf bv = *(const v16bf*)(bp + g * 512);
                acc[g] = __builtin_amdgcn_wmma_f32_16x16x32_bf16(
                    false, af.v, false, bv, (short)0, acc[g], false, false);
            }
        }
        __syncthreads();
    }

    // ---- epilogue: gate math fused -------------------------------------
    const int n_local = lane & 15;
    const int hi      = lane >> 4;
    const int u       = ut * 16 + n_local;
    const float bi  = bias[u];
    const float bc  = bias[1024 + u];
    const float bff = bias[2048 + u];
    const float bo  = bias[3072 + u];

#pragma unroll
    for (int r = 0; r < 8; ++r) {
        const int m  = mt * 16 + r + hi * 8;
        const int mu = m * 1024 + u;
        const float gi = acc[0][r] + bi;
        const float gc = acc[1][r] + bc;
        const float gf = acc[2][r] + bff;
        const float go = acc[3][r] + bo;
        if (layer == 0) {
            const float ih = c_state[mu] * sigm(gf);     // c1 * sig(f0)
            const float ii = sigm(gi) + gc;              // sig(i0) + c0 (raw)
            o0sig[mu] = sigm(go);
            a1bf[(size_t)m * 2048 + u]        = f2bf(ii);
            a1bf[(size_t)m * 2048 + 1024 + u] = f2bf(ih);
        } else {
            const float ih  = c_state[mu] * sigm(gf);    // c2 * sig(f1)
            const float ii  = sigm(gi) + tanhf(gc);
            const float nc2 = ih + ii;
            c_state[mu] = nc2;                           // new c2
            const float nh1 = tanhf(nc2) * sigm(go);
            c1_state[mu] = nh1;                          // new c1
            const float nh0 = tanhf(nh1) * o0sig[mu];
            out_t[mu] = nh0;                             // h_seq[t]
            hbf[mu]   = f2bf(nh0);                       // h for next step
        }
    }
}

// ---------------------------------------------------------------------------
extern "C" void kernel_launch(void* const* d_in, const int* in_sizes, int n_in,
                              void* d_out, int out_size, void* d_ws, size_t ws_size,
                              hipStream_t stream) {
    (void)in_sizes; (void)n_in; (void)out_size; (void)ws_size;
    const float* x    = (const float*)d_in[0];   // (512,64,1024)
    const float* k0   = (const float*)d_in[1];   // (2048,4096)
    const float* k1   = (const float*)d_in[2];   // (2048,4096)
    const float* b0   = (const float*)d_in[3];   // (4096,)
    const float* b1   = (const float*)d_in[4];   // (4096,)
    const float* h0   = (const float*)d_in[5];   // (64,1024)
    const float* c1in = (const float*)d_in[6];   // (64,1024)
    const float* c2in = (const float*)d_in[7];   // (64,1024)
    float* out = (float*)d_out;                  // (512,64,1024)

    char* ws = (char*)d_ws;
    const size_t WPK = (size_t)8388608;          // packed weight elements/layer
    unsigned short* w0p = (unsigned short*)(ws);
    unsigned short* w1p = (unsigned short*)(ws + (size_t)16 * 1024 * 1024);
    unsigned short* xbf = (unsigned short*)(ws + (size_t)32 * 1024 * 1024);
    char* p = ws + (size_t)96 * 1024 * 1024;
    unsigned short* hbf  = (unsigned short*)p;  p += 256 * 1024;
    unsigned short* a1bf = (unsigned short*)p;  p += 512 * 1024;
    float* c1s  = (float*)p;  p += 256 * 1024;
    float* c2s  = (float*)p;  p += 256 * 1024;
    float* o0s  = (float*)p;  p += 256 * 1024;

    // one-time (per call) conversions — deterministic every call
    f2bf_copy_kernel<<<4096, 256, 0, stream>>>(x, xbf, 512 * 64 * 1024);
    f2bf_copy_kernel<<<256, 256, 0, stream>>>(h0, hbf, 64 * 1024);
    fcopy_kernel<<<256, 256, 0, stream>>>(c1in, c1s, 64 * 1024);
    fcopy_kernel<<<256, 256, 0, stream>>>(c2in, c2s, 64 * 1024);
    pack_weights_kernel<<<4096, 256, 0, stream>>>(k0, w0p, (int)WPK);
    pack_weights_kernel<<<4096, 256, 0, stream>>>(k1, w1p, (int)WPK);

    for (int t = 0; t < 512; ++t) {
        // layer 0: A = [x_t | h]
        lstm_gemm_step<<<32, 256, 0, stream>>>(
            w0p, xbf + (size_t)t * 64 * 1024, hbf, 1024, 1024,
            b0, c1s, o0s, a1bf, nullptr, nullptr, nullptr, 0);
        // layer 1: A = [ii | ih] (row stride 2048)
        lstm_gemm_step<<<32, 256, 0, stream>>>(
            w1p, a1bf, a1bf + 1024, 2048, 2048,
            b1, c2s, o0s, nullptr, c1s, out + (size_t)t * 64 * 1024, hbf, 1);
    }
}
